// DWAModel_69612829934245
// MI455X (gfx1250) — matrix operand order repaired and checked by reading, not compiled, not run on hardware
//
#include <hip/hip_runtime.h>
#include <hip/hip_bf16.h>

typedef __attribute__((ext_vector_type(16))) _Float16 v16h;
typedef __attribute__((ext_vector_type(8)))  float    v8f;
typedef __attribute__((ext_vector_type(4)))  unsigned int u32x4;
typedef __attribute__((ext_vector_type(8)))  int      i32x8;
typedef __attribute__((ext_vector_type(4)))  int      i32x4;

#define BDIM 256

__device__ __forceinline__ float gelu_f(float x) {
    float x3 = x * x * x;
    return 0.5f * x * (1.f + tanhf(0.7978845608028654f * (x + 0.044715f * x3)));
}

// ---------------------------------------------------------------------------
// Tiled WMMA GEMM: C[M,N] = epilogue(A[M,K] @ B) with optional B transpose
// (B stored [N,K]), bias add, tanh-GELU, residual add. f32 -> f16 convert on
// the LDS staging path; v_wmma_f32_16x16x32_f16 with f32 accumulate.
// Block = 256 threads (8 wave32). Block tile 128x64, K step 32.
// Wave w: rows (w>>1)*32..+31, cols (w&1)*32..+31  -> 2x2 WMMA tiles/wave,
// A and B fragments each reused twice (4 WMMAs per wave per K-step).
// ---------------------------------------------------------------------------
template<bool TRANS_B, bool HAS_BIAS, bool DO_GELU, bool HAS_RES>
__global__ __launch_bounds__(BDIM)
void gemm_wmma_kernel(const float* __restrict__ A, const float* __restrict__ Bm,
                      const float* __restrict__ bias, const float* __restrict__ Res,
                      float* __restrict__ C, int M, int N, int K,
                      long long sA, long long sB, long long sC)
{
    __shared__ __align__(16) _Float16 As[128][32];  // [m][k]
    __shared__ __align__(16) _Float16 Bs[64][32];   // [n][k]  (K-transposed tile)

    const int bz = blockIdx.z;
    A  += (size_t)bz * sA;
    Bm += (size_t)bz * sB;
    C  += (size_t)bz * sC;
    const float* Rp = HAS_RES ? (Res + (size_t)bz * sC) : nullptr;

    const int m0   = blockIdx.y * 128;
    const int n0   = blockIdx.x * 64;
    const int tid  = threadIdx.x;
    const int lane = tid & 31;
    const int wid  = tid >> 5;
    const int wm   = (wid >> 1) * 32;    // wave row base (2 sub-tiles of 16)
    const int wn   = (wid & 1) * 32;     // wave col base (2 sub-tiles of 16)
    const int hlf  = lane >> 4;
    const int l15  = lane & 15;

    // global tile loader indices
    const int ar  = tid >> 1;            // 0..127 (A row), 16 floats/thread
    const int ac  = (tid & 1) * 16;      // 0,16
    const int br  = tid >> 2;            // 0..63  (B row, TRANS_B path)
    const int bc  = (tid & 3) * 8;       // 0,8,16,24
    const int bkr = tid >> 3;            // 0..31  (k row, !TRANS_B path)
    const int bnc = (tid & 7) * 8;       // n col,  !TRANS_B path

    v8f acc[2][2] = {{{}, {}}, {{}, {}}};

    for (int k0 = 0; k0 < K; k0 += 32) {
        // ---- stage A tile (convert f32 -> f16) ----
        {
            const float* src = A + (size_t)(m0 + ar) * K + (k0 + ac);
            if (k0 + 32 < K) __builtin_prefetch(src + 32, 0, 3);
            #pragma unroll
            for (int j = 0; j < 16; ++j) As[ar][ac + j] = (_Float16)src[j];
        }
        // ---- stage B tile as [n][k] ----
        if (TRANS_B) {
            const float* src = Bm + (size_t)(n0 + br) * K + (k0 + bc);
            #pragma unroll
            for (int j = 0; j < 8; ++j) Bs[br][bc + j] = (_Float16)src[j];
        } else {
            const float* src = Bm + (size_t)(k0 + bkr) * N + (n0 + bnc);
            if (k0 + 32 < K) __builtin_prefetch(src + (size_t)32 * N, 0, 3);
            #pragma unroll
            for (int j = 0; j < 8; ++j) Bs[bnc + j][bkr] = (_Float16)src[j];
        }
        __syncthreads();

        // ---- build fragments per CDNA5 WMMA VGPR layout ----
        // A: lane(m=l15), elems e<8 -> k = hlf*8+e ; e>=8 -> k = 16+hlf*8+(e-8)
        // B: lane(n=l15), elems e<8 -> k = hlf*8+e ; e>=8 -> k = 16+hlf*8+(e-8)
        union frag { v16h v; float4 f4[2]; };
        frag af[2], bf[2];
        #pragma unroll
        for (int sm = 0; sm < 2; ++sm) {
            const _Float16* arow = &As[wm + sm * 16 + l15][0];
            af[sm].f4[0] = *(const float4*)(arow + hlf * 8);
            af[sm].f4[1] = *(const float4*)(arow + 16 + hlf * 8);
        }
        #pragma unroll
        for (int sn = 0; sn < 2; ++sn) {
            const _Float16* brow = &Bs[wn + sn * 16 + l15][0];
            bf[sn].f4[0] = *(const float4*)(brow + hlf * 8);
            bf[sn].f4[1] = *(const float4*)(brow + 16 + hlf * 8);
        }

        #pragma unroll
        for (int sm = 0; sm < 2; ++sm)
            #pragma unroll
            for (int sn = 0; sn < 2; ++sn)
                acc[sm][sn] = __builtin_amdgcn_wmma_f32_16x16x32_f16(
                    false, af[sm].v, false, bf[sn].v, (short)0,
                    acc[sm][sn], false, false);
        __syncthreads();
    }

    // ---- epilogue: C/D layout lane l, vgpr g -> row = hlf*8+g, col = l15 ----
    #pragma unroll
    for (int sm = 0; sm < 2; ++sm) {
        const int rbase = m0 + wm + sm * 16 + hlf * 8;
        #pragma unroll
        for (int g = 0; g < 8; ++g) {
            const int r = rbase + g;
            #pragma unroll
            for (int sn = 0; sn < 2; ++sn) {
                const int c = n0 + wn + sn * 16 + l15;
                float v = acc[sm][sn][g];
                if (HAS_BIAS) v += bias[c];
                if (DO_GELU)  v = gelu_f(v);
                if (HAS_RES)  v += Rp[(size_t)r * N + c];
                C[(size_t)r * N + c] = v;
            }
        }
    }
}

template<bool TB, bool BIAS, bool GELUE, bool RES>
static inline void launch_gemm(const float* A, const float* Bm, const float* bias,
                               const float* res, float* C, int M, int N, int K,
                               int batch, long long sA, long long sB, long long sC,
                               hipStream_t s)
{
    dim3 grid((unsigned)(N / 64), (unsigned)(M / 128), (unsigned)batch);
    gemm_wmma_kernel<TB, BIAS, GELUE, RES><<<grid, BDIM, 0, s>>>(
        A, Bm, bias, res, C, M, N, K, sA, sB, sC);
}

// ---------------------------------------------------------------------------
// Embedding + sinusoidal positional encoding
// ---------------------------------------------------------------------------
__global__ __launch_bounds__(BDIM)
void embed_pos_kernel(const int* __restrict__ ids, const float* __restrict__ table,
                      float* __restrict__ out, int T, int D)
{
    size_t idx = (size_t)blockIdx.x * BDIM + threadIdx.x;   // over B*T*D
    int d = (int)(idx % D);
    size_t bt = idx / D;
    int t = (int)(bt % T);
    int id = ids[bt];
    int i  = (d < D / 2) ? d : (d - D / 2);
    float ang = (float)t * __powf(10000.f, -(2.f * (float)i) / (float)D);
    float pe  = (d < D / 2) ? __sinf(ang) : __cosf(ang);
    out[idx] = table[(size_t)id * D + d] + pe;
}

// ---------------------------------------------------------------------------
// LayerNorm: one block per row
// ---------------------------------------------------------------------------
__global__ __launch_bounds__(BDIM)
void layernorm_kernel(const float* __restrict__ in, float* __restrict__ out,
                      const float* __restrict__ sc, const float* __restrict__ bi, int D)
{
    __shared__ float r1[BDIM];
    __shared__ float r2[BDIM];
    const size_t row = blockIdx.x;
    const float* x = in + row * D;
    float s = 0.f, q = 0.f;
    for (int i = threadIdx.x; i < D; i += BDIM) { float v = x[i]; s += v; q += v * v; }
    r1[threadIdx.x] = s; r2[threadIdx.x] = q;
    __syncthreads();
    for (int off = BDIM / 2; off > 0; off >>= 1) {
        if ((int)threadIdx.x < off) {
            r1[threadIdx.x] += r1[threadIdx.x + off];
            r2[threadIdx.x] += r2[threadIdx.x + off];
        }
        __syncthreads();
    }
    float mean = r1[0] / (float)D;
    float var  = r2[0] / (float)D - mean * mean;
    float inv  = rsqrtf(var + 1e-6f);
    float* y = out + row * D;
    for (int i = threadIdx.x; i < D; i += BDIM)
        y[i] = (x[i] - mean) * inv * sc[i] + bi[i];
}

// ---------------------------------------------------------------------------
// Causal attention, wave32 per query row, online softmax. qkv: [B,T,3*dm],
// head h at cols h*hd (+dm for K, +2dm for V). Each lane owns 2 of hd=64 dims.
// ---------------------------------------------------------------------------
__global__ __launch_bounds__(BDIM)
void attention_kernel(const float* __restrict__ qkv, float* __restrict__ out,
                      int T, int H, int hd, int dm)
{
    const int wid = threadIdx.x >> 5, lane = threadIdx.x & 31;
    const int gid = blockIdx.x * (BDIM / 32) + wid;      // over B*H*T
    const int q = gid % T;
    const int h = (gid / T) % H;
    const int b = gid / (T * H);
    const int c0 = lane * 2;
    const int d3 = 3 * dm;
    const float* base = qkv + (size_t)b * T * d3;
    const float* qp = base + (size_t)q * d3 + h * hd;
    const float q0 = qp[c0], q1 = qp[c0 + 1];
    const float scale = rsqrtf((float)hd);
    float m = -3.4e38f, l = 0.f, o0 = 0.f, o1 = 0.f;
    for (int tk = 0; tk <= q; ++tk) {
        const float* kp = base + (size_t)tk * d3 + dm + h * hd;
        const float* vp = base + (size_t)tk * d3 + 2 * dm + h * hd;
        float p = q0 * kp[c0] + q1 * kp[c0 + 1];
        #pragma unroll
        for (int off = 16; off > 0; off >>= 1) p += __shfl_xor(p, off);
        p *= scale;
        float nm = fmaxf(m, p);
        float eo = __expf(m - nm);
        float ep = __expf(p - nm);
        l  = l * eo + ep;
        o0 = o0 * eo + ep * vp[c0];
        o1 = o1 * eo + ep * vp[c0 + 1];
        m = nm;
    }
    float* op = out + ((size_t)(b * T + q)) * dm + h * hd;
    op[c0] = o0 / l; op[c0 + 1] = o1 / l;
}

// ---------------------------------------------------------------------------
// Mean pool over T: z[b,d] = mean_t X[b,t,d]
// ---------------------------------------------------------------------------
__global__ __launch_bounds__(BDIM)
void meanpool_kernel(const float* __restrict__ X, float* __restrict__ Z, int T, int D)
{
    size_t idx = (size_t)blockIdx.x * BDIM + threadIdx.x;  // over B*D
    int d = (int)(idx % D);
    int b = (int)(idx / D);
    float s = 0.f;
    const float* p = X + ((size_t)b * T) * D + d;
    for (int t = 0; t < T; ++t) s += p[(size_t)t * D];
    Z[idx] = s / (float)T;
}

// qb[b,j] = sum_a z[b,a] * Wq[a,j]
__global__ __launch_bounds__(BDIM)
void qproj_kernel(const float* __restrict__ Z, const float* __restrict__ Wq,
                  float* __restrict__ QB, int DA, int DK)
{
    int idx = blockIdx.x * BDIM + threadIdx.x;   // over B*DK
    int j = idx % DK, b = idx / DK;
    float s = 0.f;
    for (int a = 0; a < DA; ++a) s += Z[(size_t)b * DA + a] * Wq[(size_t)a * DK + j];
    QB[idx] = s;
}

// scores[b,n] = dot(qb[b,:], pk[n,:])
__global__ __launch_bounds__(BDIM)
void scores_kernel(const float* __restrict__ QB, const float* __restrict__ PK,
                   float* __restrict__ SC, int N, int DK)
{
    int idx = blockIdx.x * BDIM + threadIdx.x;   // over B*N
    int n = idx % N, b = idx / N;
    float s = 0.f;
    for (int j = 0; j < DK; ++j) s += QB[(size_t)b * DK + j] * PK[(size_t)n * DK + j];
    SC[idx] = s;
}

// ---------------------------------------------------------------------------
// Top-K (K=16) + softmax(vals/lambda). One block per batch row.
// The 32KB scores row is staged to LDS with the Tensor Data Mover (6-arg
// amdgpu-toolchain builtin); D# describes a 1 x N tile of 4-byte elements
// landing at LDS offset 0, then s_wait_tensorcnt 0 + barrier.
// ---------------------------------------------------------------------------
__global__ __launch_bounds__(BDIM)
void topk_kernel(const float* __restrict__ SC, const float* __restrict__ lam,
                 int* __restrict__ idx_out, float* __restrict__ alpha_out,
                 int N, int KK)
{
    __shared__ float sc[8192];
    __shared__ float rv[BDIM];
    __shared__ int   ri[BDIM];
    __shared__ float vals[16];
    __shared__ int   inds[16];
    const int b = blockIdx.x;

#if __has_builtin(__builtin_amdgcn_tensor_load_to_lds) && \
    __has_builtin(__builtin_amdgcn_s_wait_tensorcnt)
    if (threadIdx.x < 32) {                 // one wave issues the TDM op
        unsigned long long ga = (unsigned long long)(SC + (size_t)b * N);
        u32x4 g0;
        g0.x = 1u;                                       // count=1 valid user D#
        g0.y = 0u;                                       // lds_addr = 0
        g0.z = (unsigned)(ga & 0xffffffffu);             // global_addr[31:0]
        g0.w = (unsigned)((ga >> 32) & 0x01ffffffu)      // global_addr[56:32]
             | (2u << 30);                               // type = 2 (image)
        i32x8 g1;
        g1[0] = 0x20000;                                 // data_size=2 (4B)
        g1[1] = (int)((unsigned)(N & 0xffff) << 16);     // tensor_dim0 lo16
        g1[2] = (int)(((unsigned)N >> 16) & 0xffffu) | (1 << 16); // dim0 hi / dim1=1
        g1[3] = (int)((unsigned)(N & 0xffff) << 16);     // tile_dim0 = N
        g1[4] = 1;                                       // tile_dim1 = 1
        g1[5] = N;                                       // tensor_dim0_stride
        g1[6] = 0;
        g1[7] = 0;
        i32x4 g2 = {0, 0, 0, 0};
        i32x4 g3 = {0, 0, 0, 0};
        i32x8 g4 = {0, 0, 0, 0, 0, 0, 0, 0};
        __builtin_amdgcn_tensor_load_to_lds(g0, g1, g2, g3, g4, 0);
        __builtin_amdgcn_s_wait_tensorcnt(0);
    }
    __syncthreads();
#else
    for (int i = threadIdx.x; i < N; i += BDIM) sc[i] = SC[(size_t)b * N + i];
    __syncthreads();
#endif

    for (int k = 0; k < KK; ++k) {
        float bm = -3.4e38f; int bi = 0;
        for (int i = threadIdx.x; i < N; i += BDIM) {
            float v = sc[i];
            if (v > bm) { bm = v; bi = i; }
        }
        rv[threadIdx.x] = bm; ri[threadIdx.x] = bi;
        __syncthreads();
        for (int off = BDIM / 2; off > 0; off >>= 1) {
            if ((int)threadIdx.x < off) {
                bool take = (rv[threadIdx.x + off] > rv[threadIdx.x]) ||
                            (rv[threadIdx.x + off] == rv[threadIdx.x] &&
                             ri[threadIdx.x + off] < ri[threadIdx.x]);
                if (take) { rv[threadIdx.x] = rv[threadIdx.x + off];
                            ri[threadIdx.x] = ri[threadIdx.x + off]; }
            }
            __syncthreads();
        }
        if (threadIdx.x == 0) { vals[k] = rv[0]; inds[k] = ri[0]; sc[ri[0]] = -3.4e38f; }
        __syncthreads();
    }
    if (threadIdx.x == 0) {
        float L = lam[0];
        float mx = vals[0];                     // sorted descending
        float e[16], se = 0.f;
        for (int k = 0; k < KK; ++k) { e[k] = __expf((vals[k] - mx) / L); se += e[k]; }
        for (int k = 0; k < KK; ++k) {
            alpha_out[b * KK + k] = e[k] / se;
            idx_out[b * KK + k] = inds[k];
        }
    }
}

// ---------------------------------------------------------------------------
// W[b,d,a] = W_base[d,a] + gamma * sum_k alpha[b,k] * sum_r U[d,r]*V[r,a]
//   gathered row g = pool[idx]; U[d,r]=g[d*R+r]; V[r,a]=g[DB*R + r*DA + a]
// ---------------------------------------------------------------------------
__global__ __launch_bounds__(BDIM)
void assemble_w_kernel(const float* __restrict__ pool, const int* __restrict__ idx,
                       const float* __restrict__ alpha, const float* __restrict__ Wbase,
                       const float* __restrict__ gamma, float* __restrict__ W,
                       int DA, int DB, int R, int KK, int DP)
{
    size_t id = (size_t)blockIdx.x * BDIM + threadIdx.x;   // over B*DB*DA
    int a = (int)(id % DA);
    size_t t = id / DA;
    int d = (int)(t % DB);
    int b = (int)(t / DB);
    float acc = 0.f;
    for (int k = 0; k < KK; ++k) {
        const float* g = pool + (size_t)idx[b * KK + k] * DP;
        float al = alpha[b * KK + k];
        float s = 0.f;
        #pragma unroll
        for (int r = 0; r < 4; ++r)
            s += g[d * R + r] * g[(size_t)DB * R + (size_t)r * DA + a];
        acc += al * s;
    }
    W[id] = Wbase[(size_t)d * DA + a] + gamma[0] * acc;
}

// ---------------------------------------------------------------------------
// Host-side driver
// ---------------------------------------------------------------------------
static void run_layer(const float* const p[10], int li,
                      float* X, float* T1, float* T2, float* T3, hipStream_t s)
{
    const int M = 4096, D = 1024;
    // attention block
    layernorm_kernel<<<M, BDIM, 0, s>>>(X, T2, p[0] + (size_t)li * D, p[1] + (size_t)li * D, D);
    launch_gemm<false, false, false, false>(T2, p[2] + (size_t)li * D * 3 * D, nullptr, nullptr,
                                            T1, M, 3 * D, D, 1, 0, 0, 0, s);
    attention_kernel<<<(4 * 16 * 1024) / (BDIM / 32), BDIM, 0, s>>>(T1, T3, 1024, 16, 64, D);
    launch_gemm<false, false, false, true>(T3, p[3] + (size_t)li * D * D, nullptr, X,
                                           X, M, D, D, 1, 0, 0, 0, s);
    // FFN block
    layernorm_kernel<<<M, BDIM, 0, s>>>(X, T2, p[4] + (size_t)li * D, p[5] + (size_t)li * D, D);
    launch_gemm<false, true, true, false>(T2, p[6] + (size_t)li * D * 4 * D,
                                          p[7] + (size_t)li * 4 * D, nullptr,
                                          T1, M, 4 * D, D, 1, 0, 0, 0, s);
    launch_gemm<false, true, false, true>(T1, p[8] + (size_t)li * 4 * D * D,
                                          p[9] + (size_t)li * D, X,
                                          X, M, D, 4 * D, 1, 0, 0, 0, s);
}

extern "C" void kernel_launch(void* const* d_in, const int* in_sizes, int n_in,
                              void* d_out, int out_size, void* d_ws, size_t ws_size,
                              hipStream_t stream)
{
    (void)in_sizes; (void)n_in; (void)out_size; (void)ws_size;
    const int   B = 4, T = 1024, V = 32000, DA = 1024, DB = 1024, R = 4;
    const int   N = 8192, DK = 128, TOPK = 16, DP = R * (DA + DB);

    const int*   input_ids  = (const int*)d_in[0];
    const float* lambda_val = (const float*)d_in[1];
    const float* embed_tab  = (const float*)d_in[2];
    const float* pa[10]; for (int i = 0; i < 10; ++i) pa[i] = (const float*)d_in[3 + i];
    const float* pool       = (const float*)d_in[13];
    const float* key_proj   = (const float*)d_in[14];
    const float* query_proj = (const float*)d_in[15];
    const float* W_base     = (const float*)d_in[16];
    const float* b_base     = (const float*)d_in[17];
    const float* gamma      = (const float*)d_in[18];
    const float* asm_s      = (const float*)d_in[19];
    const float* asm_b      = (const float*)d_in[20];
    const float* pb[10]; for (int i = 0; i < 10; ++i) pb[i] = (const float*)d_in[21 + i];
    const float* lm_head    = (const float*)d_in[31];
    float*       out        = (float*)d_out;

    float* w = (float*)d_ws;
    size_t off = 0;
    auto alloc = [&](size_t n) { float* p = w + off; off += n; return p; };
    float* X  = alloc((size_t)B * T * DA);        // residual stream / h_A
    float* T1 = alloc((size_t)B * T * 4 * DA);    // qkv / ffn hidden
    float* T2 = alloc((size_t)B * T * DA);        // LN out / h_mid
    float* T3 = alloc((size_t)B * T * DA);        // attention out
    float* PK = alloc((size_t)N * DK);            // pool keys
    float* Z  = alloc((size_t)B * DA);
    float* QB = alloc((size_t)B * DK);
    float* SC = alloc((size_t)B * N);
    float* WB = alloc((size_t)B * DB * DA);       // assembled W
    float* AL = alloc(B * TOPK);
    int*   IX = (int*)alloc(B * TOPK);

    // 1. embed + posenc
    embed_pos_kernel<<<(B * T * DA) / BDIM, BDIM, 0, stream>>>(input_ids, embed_tab, X, T, DA);
    // 2. Part A transformer
    for (int li = 0; li < 2; ++li) run_layer(pa, li, X, T1, T2, T3, stream);
    // 3. retrieval
    meanpool_kernel<<<(B * DA) / BDIM, BDIM, 0, stream>>>(X, Z, T, DA);
    launch_gemm<false, false, false, false>(pool, key_proj, nullptr, nullptr,
                                            PK, N, DK, DP, 1, 0, 0, 0, stream);
    qproj_kernel<<<(B * DK) / BDIM, BDIM, 0, stream>>>(Z, query_proj, QB, DA, DK);
    scores_kernel<<<(B * N) / BDIM, BDIM, 0, stream>>>(QB, PK, SC, N, DK);
    topk_kernel<<<B, BDIM, 0, stream>>>(SC, lambda_val, IX, AL, N, TOPK);
    // 4. dynamic weight assembly
    assemble_w_kernel<<<((size_t)B * DB * DA) / BDIM, BDIM, 0, stream>>>(
        pool, IX, AL, W_base, gamma, WB, DA, DB, R, TOPK, DP);
    // 5. h_mid = h_A @ W[b]^T + b_base  (batched, B transposed), then LN
    launch_gemm<true, true, false, false>(X, WB, b_base, nullptr, T2,
                                          T, DB, DA, B,
                                          (long long)T * DA, (long long)DB * DA,
                                          (long long)T * DB, stream);
    layernorm_kernel<<<B * T, BDIM, 0, stream>>>(T2, X, asm_s, asm_b, DB);
    // 6. Part B transformer
    for (int li = 0; li < 2; ++li) run_layer(pb, li, X, T1, T2, T3, stream);
    // 7. LM head -> logits
    launch_gemm<false, false, false, false>(X, lm_head, nullptr, nullptr,
                                            out, B * T, V, DB, 1, 0, 0, 0, stream);
}
